// TransformerAttention_5574867551017
// MI455X (gfx1250) — matrix operand order repaired and checked by reading, not compiled, not run on hardware
//
#include <hip/hip_runtime.h>

#define S_LEN 4096
#define D_DIM 768
#define NHEAD 12
#define DHEAD 64
#define BATCH 2
#define QK_SCALE 0.125f   // 1/sqrt(64)

typedef __attribute__((ext_vector_type(16))) __bf16 v16bf;
typedef __attribute__((ext_vector_type(8)))  float  v8f;

__device__ __forceinline__ v8f wmma_bf16(v16bf a, v16bf b, v8f c) {
  // D = A(16x32 bf16) x B(32x16 bf16) + C(16x16 f32)
  return __builtin_amdgcn_wmma_f32_16x16x32_bf16(false, a, false, b, (short)0, c,
                                                 false, false);
}

// Load 16 bf16 from two 8-element (16B) segments -> one fragment register set.
__device__ __forceinline__ v16bf ld_bf16_2seg(const __bf16* p0, const __bf16* p1) {
  union { v16bf v; uint4 u[2]; } r;
  r.u[0] = *(const uint4*)p0;
  r.u[1] = *(const uint4*)p1;
  return r.v;
}
__device__ __forceinline__ v16bf ld_bf16_16(const __bf16* p) {
  return ld_bf16_2seg(p, p + 8);
}

// ---------------------------------------------------------------------------
// Kernel 0: one-shot f32 -> bf16 conversion (activations + weights).
// ---------------------------------------------------------------------------
__global__ void __launch_bounds__(256) f32_to_bf16_kernel(
    const float* __restrict__ in, __bf16* __restrict__ out, int n4) {
  const int i = blockIdx.x * 256 + threadIdx.x;
  if (i < n4) {
    const float4 f = ((const float4*)in)[i];
    union { __bf16 h[4]; uint2 u; } r;
    r.h[0] = (__bf16)f.x;
    r.h[1] = (__bf16)f.y;
    r.h[2] = (__bf16)f.z;
    r.h[3] = (__bf16)f.w;
    ((uint2*)out)[i] = r.u;
  }
}

// ---------------------------------------------------------------------------
// Kernel 1: fused QKV projection.  out = x @ W^T + b  (per matrix), bf16 I/O.
// Q,K stored [B,H,S,DH]; V stored transposed [B,H,DH,S].
// Double-buffered A/B fragments: next k-step loads issued before current
// fragments are consumed, keeping global loads in flight under the WMMAs.
// ---------------------------------------------------------------------------
__global__ void __launch_bounds__(128) qkv_proj_kernel(
    const __bf16* __restrict__ xb,
    const __bf16* __restrict__ Wqb, const float* __restrict__ bq,
    const __bf16* __restrict__ Wkb, const float* __restrict__ bk,
    const __bf16* __restrict__ Wvb, const float* __restrict__ bv,
    __bf16* __restrict__ qbuf, __bf16* __restrict__ kbuf,
    __bf16* __restrict__ vbuf) {
  const int wid  = blockIdx.x * 4 + (threadIdx.x >> 5);
  const int lane = threadIdx.x & 31;
  const int hw   = lane >> 4;   // half-wave select
  const int l15  = lane & 15;

  const int mat   = wid / (NHEAD * 512);
  const int rem   = wid % (NHEAD * 512);
  const int hd    = rem / 512;        // head / 64-col group
  const int mtile = rem % 512;
  const int m0    = mtile * 16;

  const __bf16* W    = (mat == 0) ? Wqb : (mat == 1) ? Wkb : Wvb;
  const float*  bias = (mat == 0) ? bq : (mat == 1) ? bk : bv;

  v8f acc[4] = {};
  const __bf16* xrow  = xb + (size_t)(m0 + l15) * D_DIM;          // A: row = l15
  const __bf16* wbase = W + (size_t)(hd * DHEAD + l15) * D_DIM;   // B: n-row = l15

  v16bf a = ld_bf16_16(xrow + hw * 16);
  v16bf b0 = ld_bf16_16(wbase + (size_t)0 * 16 * D_DIM + hw * 16);
  v16bf b1 = ld_bf16_16(wbase + (size_t)1 * 16 * D_DIM + hw * 16);
  v16bf b2 = ld_bf16_16(wbase + (size_t)2 * 16 * D_DIM + hw * 16);
  v16bf b3 = ld_bf16_16(wbase + (size_t)3 * 16 * D_DIM + hw * 16);

  for (int k0 = 0; k0 < D_DIM; k0 += 32) {
    const int k1 = (k0 + 32) % D_DIM;   // wraps to 0 on last iter (unused data)
    // issue next-step loads first so they age behind the WMMAs
    v16bf na  = ld_bf16_16(xrow + k1 + hw * 16);
    v16bf nb0 = ld_bf16_16(wbase + (size_t)0 * 16 * D_DIM + k1 + hw * 16);
    v16bf nb1 = ld_bf16_16(wbase + (size_t)1 * 16 * D_DIM + k1 + hw * 16);
    v16bf nb2 = ld_bf16_16(wbase + (size_t)2 * 16 * D_DIM + k1 + hw * 16);
    v16bf nb3 = ld_bf16_16(wbase + (size_t)3 * 16 * D_DIM + k1 + hw * 16);
    acc[0] = wmma_bf16(a, b0, acc[0]);
    acc[1] = wmma_bf16(a, b1, acc[1]);
    acc[2] = wmma_bf16(a, b2, acc[2]);
    acc[3] = wmma_bf16(a, b3, acc[3]);
    a = na; b0 = nb0; b1 = nb1; b2 = nb2; b3 = nb3;
  }

#pragma unroll
  for (int nt = 0; nt < 4; ++nt) {
    const int dh   = nt * 16 + l15;
    const int ncol = hd * DHEAD + dh;
    const float bval = bias[ncol];
#pragma unroll
    for (int r = 0; r < 8; ++r) {
      const int m    = m0 + hw * 8 + r;
      const int bidx = m >> 12;            // m / 4096
      const int s    = m & (S_LEN - 1);
      float v = acc[nt][r] + bval;
      if (mat == 0) {
        v *= QK_SCALE;
        qbuf[((size_t)(bidx * NHEAD + hd) * S_LEN + s) * DHEAD + dh] = (__bf16)v;
      } else if (mat == 1) {
        kbuf[((size_t)(bidx * NHEAD + hd) * S_LEN + s) * DHEAD + dh] = (__bf16)v;
      } else {
        vbuf[((size_t)(bidx * NHEAD + hd) * DHEAD + dh) * S_LEN + s] = (__bf16)v;
      }
    }
  }
}

// ---------------------------------------------------------------------------
// Kernel 2: flash attention.  One wave owns 16 query rows of one (b,h).
// KV step = 32 keys.  Software-pipelined: current-step V loads and next-step
// K loads are issued immediately after the score WMMAs, so all global traffic
// is in flight while the softmax VALU section and LDS P-relayout execute.
// ---------------------------------------------------------------------------
__global__ void __launch_bounds__(128) flash_attn_kernel(
    const __bf16* __restrict__ qbuf, const __bf16* __restrict__ kbuf,
    const __bf16* __restrict__ vbuf, const float* __restrict__ mask,
    __bf16* __restrict__ obuf) {
  __shared__ __bf16 Plds[4][16][32];   // per-wave 16x32 P tile (1 KB each)
  const int wv   = threadIdx.x >> 5;
  const int lane = threadIdx.x & 31;
  const int hw   = lane >> 4;
  const int l15  = lane & 15;
  const int h    = blockIdx.y;
  const int bidx = blockIdx.z;
  const int q0   = (blockIdx.x * 4 + wv) * 16;

  const __bf16* qh = qbuf + (size_t)(bidx * NHEAD + h) * S_LEN * DHEAD;
  const __bf16* kh = kbuf + (size_t)(bidx * NHEAD + h) * S_LEN * DHEAD;
  const __bf16* vh = vbuf + (size_t)(bidx * NHEAD + h) * DHEAD * S_LEN;

  // Q A-fragments resident for entire KV loop (dh 0..31 and 32..63).
  const __bf16* qrow = qh + (size_t)(q0 + l15) * DHEAD;
  const v16bf qa0 = ld_bf16_16(qrow + hw * 16);
  const v16bf qa1 = ld_bf16_16(qrow + 32 + hw * 16);

  float mrow[8], lrow[8];
#pragma unroll
  for (int r = 0; r < 8; ++r) { mrow[r] = -1e30f; lrow[r] = 0.0f; }
  v8f o[4] = {};

  // Preload K fragments for step 0 (keys 0..31, dh in two 32-wide halves).
  v16bf kf0, kf1, kf2, kf3;
  {
    const __bf16* kr0 = kh + (size_t)l15 * DHEAD;
    const __bf16* kr1 = kh + (size_t)(16 + l15) * DHEAD;
    kf0 = ld_bf16_16(kr0 + hw * 16);
    kf1 = ld_bf16_16(kr0 + 32 + hw * 16);
    kf2 = ld_bf16_16(kr1 + hw * 16);
    kf3 = ld_bf16_16(kr1 + 32 + hw * 16);
  }

  for (int kb0 = 0; kb0 < S_LEN; kb0 += 32) {
    // ---- scores: S = (Q*scale) @ K^T for 32 keys (two 16-col fragments)
    v8f s0 = {}, s1 = {};
    s0 = wmma_bf16(qa0, kf0, s0);
    s0 = wmma_bf16(qa1, kf1, s0);
    s1 = wmma_bf16(qa0, kf2, s1);
    s1 = wmma_bf16(qa1, kf3, s1);

    // ---- issue current-step V loads (consumed after softmax)
    v16bf vf0 = ld_bf16_16(vh + (size_t)(0 * 16 + l15) * S_LEN + kb0 + hw * 16);
    v16bf vf1 = ld_bf16_16(vh + (size_t)(1 * 16 + l15) * S_LEN + kb0 + hw * 16);
    v16bf vf2 = ld_bf16_16(vh + (size_t)(2 * 16 + l15) * S_LEN + kb0 + hw * 16);
    v16bf vf3 = ld_bf16_16(vh + (size_t)(3 * 16 + l15) * S_LEN + kb0 + hw * 16);

    // ---- issue next-step K loads (wrapped on last iter; data unused then)
    const int kn = (kb0 + 32) & (S_LEN - 1);
    const __bf16* nr0 = kh + (size_t)(kn + l15) * DHEAD;
    const __bf16* nr1 = kh + (size_t)(kn + 16 + l15) * DHEAD;
    v16bf nk0 = ld_bf16_16(nr0 + hw * 16);
    v16bf nk1 = ld_bf16_16(nr0 + 32 + hw * 16);
    v16bf nk2 = ld_bf16_16(nr1 + hw * 16);
    v16bf nk3 = ld_bf16_16(nr1 + 32 + hw * 16);

    // ---- additive attention mask [S,S]
#pragma unroll
    for (int r = 0; r < 8; ++r) {
      const size_t qs = (size_t)(q0 + hw * 8 + r) * S_LEN;
      s0[r] += mask[qs + kb0 + l15];
      s1[r] += mask[qs + kb0 + 16 + l15];
    }

    // ---- online softmax (row-wise over the 16 lanes of each half-wave)
#pragma unroll
    for (int r = 0; r < 8; ++r) {
      float rmax = fmaxf(s0[r], s1[r]);
      rmax = fmaxf(rmax, __shfl_xor(rmax, 8, 16));
      rmax = fmaxf(rmax, __shfl_xor(rmax, 4, 16));
      rmax = fmaxf(rmax, __shfl_xor(rmax, 2, 16));
      rmax = fmaxf(rmax, __shfl_xor(rmax, 1, 16));
      const float mnew = fmaxf(mrow[r], rmax);
      const float sc   = __expf(mrow[r] - mnew);
      const float p0   = __expf(s0[r] - mnew);
      const float p1   = __expf(s1[r] - mnew);
      float rsum = p0 + p1;
      rsum += __shfl_xor(rsum, 8, 16);
      rsum += __shfl_xor(rsum, 4, 16);
      rsum += __shfl_xor(rsum, 2, 16);
      rsum += __shfl_xor(rsum, 1, 16);
      lrow[r] = lrow[r] * sc + rsum;
      mrow[r] = mnew;
      o[0][r] *= sc; o[1][r] *= sc; o[2][r] *= sc; o[3][r] *= sc;
      const int m = hw * 8 + r;                       // C-fragment row
      Plds[wv][m][l15]      = (__bf16)p0;
      Plds[wv][m][16 + l15] = (__bf16)p1;
    }
    asm volatile("s_wait_dscnt 0" ::: "memory");      // P stores visible to wave

    // ---- O += P @ V  (A = P 16x32 from LDS, B = V^T fragments)
    const __bf16* prow = &Plds[wv][l15][0];
    const v16bf pa = ld_bf16_16(prow + hw * 16);
    o[0] = wmma_bf16(pa, vf0, o[0]);
    o[1] = wmma_bf16(pa, vf1, o[1]);
    o[2] = wmma_bf16(pa, vf2, o[2]);
    o[3] = wmma_bf16(pa, vf3, o[3]);

    kf0 = nk0; kf1 = nk1; kf2 = nk2; kf3 = nk3;
  }

  // ---- normalize and write O (bf16, [B,S,D]) for the output projection
#pragma unroll
  for (int r = 0; r < 8; ++r) {
    const float inv = 1.0f / lrow[r];
    const int s = q0 + hw * 8 + r;
#pragma unroll
    for (int nt = 0; nt < 4; ++nt) {
      const int d = h * DHEAD + nt * 16 + l15;
      obuf[((size_t)bidx * S_LEN + s) * D_DIM + d] = (__bf16)(o[nt][r] * inv);
    }
  }
}

// ---------------------------------------------------------------------------
// Kernel 3: output projection.  out = O_bf16 @ Wo_bf16^T + bo  (f32 out).
// Double-buffered like kernel 1.
// ---------------------------------------------------------------------------
__global__ void __launch_bounds__(128) out_proj_kernel(
    const __bf16* __restrict__ obuf, const __bf16* __restrict__ Wob,
    const float* __restrict__ bo, float* __restrict__ out) {
  const int wid  = blockIdx.x * 4 + (threadIdx.x >> 5);
  const int lane = threadIdx.x & 31;
  const int hw   = lane >> 4;
  const int l15  = lane & 15;
  const int ng    = wid / 512;
  const int mtile = wid % 512;
  const int m0    = mtile * 16;

  v8f acc[4] = {};
  const __bf16* orow  = obuf + (size_t)(m0 + l15) * D_DIM;
  const __bf16* wbase = Wob + (size_t)(ng * 64 + l15) * D_DIM;

  v16bf a = ld_bf16_16(orow + hw * 16);
  v16bf b0 = ld_bf16_16(wbase + (size_t)0 * 16 * D_DIM + hw * 16);
  v16bf b1 = ld_bf16_16(wbase + (size_t)1 * 16 * D_DIM + hw * 16);
  v16bf b2 = ld_bf16_16(wbase + (size_t)2 * 16 * D_DIM + hw * 16);
  v16bf b3 = ld_bf16_16(wbase + (size_t)3 * 16 * D_DIM + hw * 16);

  for (int k0 = 0; k0 < D_DIM; k0 += 32) {
    const int k1 = (k0 + 32) % D_DIM;   // wraps to 0 on last iter (unused data)
    v16bf na  = ld_bf16_16(orow + k1 + hw * 16);
    v16bf nb0 = ld_bf16_16(wbase + (size_t)0 * 16 * D_DIM + k1 + hw * 16);
    v16bf nb1 = ld_bf16_16(wbase + (size_t)1 * 16 * D_DIM + k1 + hw * 16);
    v16bf nb2 = ld_bf16_16(wbase + (size_t)2 * 16 * D_DIM + k1 + hw * 16);
    v16bf nb3 = ld_bf16_16(wbase + (size_t)3 * 16 * D_DIM + k1 + hw * 16);
    acc[0] = wmma_bf16(a, b0, acc[0]);
    acc[1] = wmma_bf16(a, b1, acc[1]);
    acc[2] = wmma_bf16(a, b2, acc[2]);
    acc[3] = wmma_bf16(a, b3, acc[3]);
    a = na; b0 = nb0; b1 = nb1; b2 = nb2; b3 = nb3;
  }

#pragma unroll
  for (int nt = 0; nt < 4; ++nt) {
    const int n = ng * 64 + nt * 16 + l15;
    const float bval = bo[n];
#pragma unroll
    for (int r = 0; r < 8; ++r) {
      const int m = m0 + hw * 8 + r;
      out[(size_t)m * D_DIM + n] = acc[nt][r] + bval;
    }
  }
}

// ---------------------------------------------------------------------------
extern "C" void kernel_launch(void* const* d_in, const int* in_sizes, int n_in,
                              void* d_out, int out_size, void* d_ws, size_t ws_size,
                              hipStream_t stream) {
  (void)in_sizes; (void)n_in; (void)out_size; (void)ws_size;
  const float* seq  = (const float*)d_in[0];
  const float* mask = (const float*)d_in[1];
  const float* Wq   = (const float*)d_in[2];
  const float* bq   = (const float*)d_in[3];
  const float* Wk   = (const float*)d_in[4];
  const float* bk   = (const float*)d_in[5];
  const float* Wv   = (const float*)d_in[6];
  const float* bv   = (const float*)d_in[7];
  const float* Wo   = (const float*)d_in[8];
  const float* bo   = (const float*)d_in[9];
  float* out = (float*)d_out;

  const size_t act_elems = (size_t)BATCH * NHEAD * S_LEN * DHEAD;  // 6,291,456
  const size_t w_elems   = (size_t)D_DIM * D_DIM;                  //   589,824
  __bf16* qb  = (__bf16*)d_ws;
  __bf16* kb  = qb + act_elems;
  __bf16* vb  = kb + act_elems;    // V stored transposed [B,H,DH,S]
  __bf16* ob  = vb + act_elems;    // attention output bf16 [B,S,D]
  __bf16* xbb = ob + act_elems;    // sequence bf16 [B*S, D]
  __bf16* wqb = xbb + act_elems;
  __bf16* wkb = wqb + w_elems;
  __bf16* wvb = wkb + w_elems;
  __bf16* wob = wvb + w_elems;

  // ---- one-shot f32 -> bf16 conversions (activations + weights)
  {
    const int n4x = (int)(act_elems / 4);   // 1,572,864
    const int n4w = (int)(w_elems / 4);     //   147,456
    f32_to_bf16_kernel<<<(n4x + 255) / 256, 256, 0, stream>>>(seq, xbb, n4x);
    f32_to_bf16_kernel<<<(n4w + 255) / 256, 256, 0, stream>>>(Wq, wqb, n4w);
    f32_to_bf16_kernel<<<(n4w + 255) / 256, 256, 0, stream>>>(Wk, wkb, n4w);
    f32_to_bf16_kernel<<<(n4w + 255) / 256, 256, 0, stream>>>(Wv, wvb, n4w);
    f32_to_bf16_kernel<<<(n4w + 255) / 256, 256, 0, stream>>>(Wo, wob, n4w);
  }

  // 18432 waves / 4 per block
  qkv_proj_kernel<<<4608, 128, 0, stream>>>(xbb, wqb, bq, wkb, bk, wvb, bv,
                                            qb, kb, vb);
  // 64 q-blocks (of 4 waves x 16 rows) per (h,b)
  flash_attn_kernel<<<dim3(64, NHEAD, BATCH), 128, 0, stream>>>(qb, kb, vb,
                                                                mask, ob);
  // 6144 waves / 4 per block
  out_proj_kernel<<<1536, 128, 0, stream>>>(ob, wob, bo, out);
}
